// _Attention_19816979103990
// MI455X (gfx1250) — matrix operand order repaired and checked by reading, not compiled, not run on hardware
//
#include <hip/hip_runtime.h>

// MI455X / gfx1250: wave32, WMMA f32 <- f16 16x16x32.
// Pipeline: [f32->f16 convert] -> [QKV gemm, async-LDS double-buffered]
//           -> [flash attention, l via ones-WMMA, permlane16 max-reduce]
//           -> [proj gemm, async-LDS double-buffered]

typedef __attribute__((ext_vector_type(16))) _Float16 v16h;
typedef __attribute__((ext_vector_type(8)))  _Float16 v8h;
typedef __attribute__((ext_vector_type(4)))  _Float16 v4h;
typedef __attribute__((ext_vector_type(8)))  float    v8f;

__device__ __forceinline__ v8f wmma_f16(v16h a, v16h b, v8f c) {
  // D = A(16x32 f16) * B(32x16 f16) + C(16x16 f32)
  return __builtin_amdgcn_wmma_f32_16x16x32_f16(false, a, false, b, (short)0, c,
                                                false, false);
}

// A-fragment (16x32 f16, M x K): lane holds row m = lane%16; g = lane/16.
// halves 0..15 = row[g*16 .. +15] (k-contiguous)
__device__ __forceinline__ v16h load_a_frag(const _Float16* row, int g) {
  v8h lo = *(const v8h*)(row + g * 16);
  v8h hi = *(const v8h*)(row + g * 16 + 8);
  v16h r;
#pragma unroll
  for (int i = 0; i < 8; ++i) { r[i] = lo[i]; r[i + 8] = hi[i]; }
  return r;
}

// B-fragment (32x16 f16, K x N): lane = column n = lane%16, K = g*16..+15
// sequential; p points at &Bcol[n][g*16] (k-contiguous).
__device__ __forceinline__ v16h load_b_frag(const _Float16* p) {
  v8h lo = *(const v8h*)(p);
  v8h hi = *(const v8h*)(p + 8);
  v16h r;
#pragma unroll
  for (int i = 0; i < 8; ++i) { r[i] = lo[i]; r[i + 8] = hi[i]; }
  return r;
}

// Async global -> LDS 16B copy (ASYNCcnt). Generic LDS addresses carry the
// LDS byte offset in the low 32 bits (flat aperture rule), so truncation is
// the DS address the instruction needs.
__device__ __forceinline__ void async_copy_b128(void* lds_dst, const void* gsrc) {
  unsigned l = (unsigned)(uintptr_t)lds_dst;
  asm volatile("global_load_async_to_lds_b128 %0, %1, off"
               :: "v"(l), "v"(gsrc) : "memory");
}
__device__ __forceinline__ void wait_async0() {
  asm volatile("s_wait_asynccnt 0x0" ::: "memory");
}

// Width-16 xor-butterfly max without touching the LDS pipe: v_permlane16
// (same row-relative swizzle applied to both 16-lane rows of the wave).
#if __has_builtin(__builtin_amdgcn_permlane16)
__device__ __forceinline__ float perml16(float v, unsigned s0, unsigned s1) {
  unsigned u = __float_as_uint(v);
  unsigned r = __builtin_amdgcn_permlane16(u, u, s0, s1, false, false);
  return __uint_as_float(r);
}
__device__ __forceinline__ float row16_max(float r) {
  r = fmaxf(r, perml16(r, 0x67452301u, 0xEFCDAB89u));  // j^1
  r = fmaxf(r, perml16(r, 0x54761032u, 0xDCFEBA98u));  // j^2 in lanes 0-7, 12-15; lanes 8-11 keep own value
  r = fmaxf(r, perml16(r, 0x32107654u, 0xBA98FEDCu));  // j^4
  r = fmaxf(r, perml16(r, 0xFEDCBA98u, 0x76543210u));  // j^8
  return r;
}
#else
__device__ __forceinline__ float row16_max(float r) {
#pragma unroll
  for (int off = 1; off < 16; off <<= 1) r = fmaxf(r, __shfl_xor(r, off, 32));
  return r;
}
#endif

// ---------------------------------------------------------------------------
// One-time f32 -> f16 conversion (x, w_qkv, w_proj). n % 4 == 0.
// ---------------------------------------------------------------------------
__global__ __launch_bounds__(256)
void cvt_f32_to_f16_kernel(const float* __restrict__ src,
                           _Float16* __restrict__ dst, int n4) {
  int i = blockIdx.x * 256 + threadIdx.x;
  if (i < n4) {
    float4 f = *(const float4*)(src + (size_t)i * 4);
    v4h h; h[0] = (_Float16)f.x; h[1] = (_Float16)f.y;
           h[2] = (_Float16)f.z; h[3] = (_Float16)f.w;
    *(v4h*)(dst + (size_t)i * 4) = h;
  }
}

// ---------------------------------------------------------------------------
// GEMM: C[M x Nout] = A[M x K] * W[Nout x K]^T + bias, all-f16 operands.
// Double-buffered async global->LDS staging (panels 128x32, pad to 40 halves:
// 20-dword stride = conflict-free b128 fragment reads, 16B-aligned chunks).
// Block 256 thr / 8 waves; tile 128x128; wave 32x64 -> 8 WMMAs per panel.
// MODE 0: scatter f16 into q/k/v [B][H][N][D];  MODE 1: f32 to d_out.
// ---------------------------------------------------------------------------
template <int MODE>
__global__ __launch_bounds__(256)
void gemm_f16_kernel(const _Float16* __restrict__ Ag,
                     const _Float16* __restrict__ Wg,
                     const float* __restrict__ bias,
                     _Float16* __restrict__ qo,
                     _Float16* __restrict__ ko,
                     _Float16* __restrict__ vo,
                     float* __restrict__ Co,
                     int K, int Nout) {
  __shared__ _Float16 As[2][128 * 40];
  __shared__ _Float16 Bs[2][128 * 40];

  const int tid = threadIdx.x;
  const int w = tid >> 5, lane = tid & 31;
  const int g = lane >> 4, ln = lane & 15;
  const int wm = w >> 1, wn = w & 1;        // 4 x 2 wave grid
  const int bm0 = blockIdx.x * 128, bn0 = blockIdx.y * 128;

  v8f c[2][4];
#pragma unroll
  for (int a = 0; a < 2; ++a)
#pragma unroll
    for (int b = 0; b < 4; ++b)
#pragma unroll
      for (int i = 0; i < 8; ++i) c[a][b][i] = 0.0f;

  // Stage panel at k-offset kk into buffer p: 4 async b128 chunks per thread.
  auto stage = [&](int p, int kk) {
#pragma unroll
    for (int r = 0; r < 2; ++r) {
      int idx = r * 2048 + tid * 8;         // 128 x 32 halves
      int row = idx >> 5, col = idx & 31;
      async_copy_b128(&As[p][row * 40 + col],
                      Ag + (size_t)(bm0 + row) * K + kk + col);
      async_copy_b128(&Bs[p][row * 40 + col],
                      Wg + (size_t)(bn0 + row) * K + kk + col);
    }
  };

  stage(0, 0);
  wait_async0();
  __syncthreads();

  int p = 0;
  for (int kk = 0; kk < K; kk += 32) {
    if (kk + 32 < K) stage(p ^ 1, kk + 32);  // overlap next panel with WMMAs

    v16h a0 = load_a_frag(&As[p][(wm * 32 + ln) * 40], g);
    v16h a1 = load_a_frag(&As[p][(wm * 32 + 16 + ln) * 40], g);
#pragma unroll
    for (int fn = 0; fn < 4; ++fn) {
      v16h bf = load_b_frag(&Bs[p][(wn * 64 + fn * 16 + ln) * 40 + g * 16]);
      c[0][fn] = wmma_f16(a0, bf, c[0][fn]);
      c[1][fn] = wmma_f16(a1, bf, c[1][fn]);
    }

    wait_async0();       // this wave's async copies landed
    __syncthreads();     // all waves: reads of p done, writes of p^1 visible
    p ^= 1;
  }

  // Epilogue. C layout: element i -> row m = g*8+i, col n = ln.
#pragma unroll
  for (int fm = 0; fm < 2; ++fm)
#pragma unroll
    for (int fn = 0; fn < 4; ++fn)
#pragma unroll
      for (int i = 0; i < 8; ++i) {
        int gm = bm0 + wm * 32 + fm * 16 + g * 8 + i;
        int gn = bn0 + wn * 64 + fn * 16 + ln;
        float v = c[fm][fn][i] + bias[gn];
        if (MODE == 0) {
          int sel = gn >> 10, cc = gn & 1023;
          int h = cc >> 6, d = cc & 63;       // head, head-dim
          int b = gm >> 11, ns = gm & 2047;   // batch, seq
          _Float16* dst = (sel == 0) ? qo : (sel == 1) ? ko : vo;
          dst[(((size_t)(b * 16 + h)) * 2048 + ns) * 64 + d] = (_Float16)v;
        } else {
          Co[(size_t)gm * Nout + gn] = v;
        }
      }
}

// ---------------------------------------------------------------------------
// Flash attention over q/k/v f16 [B][H][N=2048][D=64] (unscaled logits, as in
// the reference). Block = 256 threads (8 waves) = 128 query rows of one (b,h);
// wave owns 16 rows; 16 key tiles of 128 keys.
// Row-sum l is carried as a WMMA accumulator against an all-ones B matrix
// (every output column == row sum -> every lane holds l, no shuffles).
// ---------------------------------------------------------------------------
__global__ __launch_bounds__(256)
void flash_attn_kernel(const _Float16* __restrict__ qf,
                       const _Float16* __restrict__ kf,
                       const _Float16* __restrict__ vf,
                       _Float16* __restrict__ ao) {
  __shared__ _Float16 Pl[8 * 16 * 136];  // per-wave 16x128 P (68-dword stride)
  __shared__ _Float16 Vt[64 * 136];      // V^T tile: [d=64][n=128]

  const int tid = threadIdx.x;
  const int w = tid >> 5, lane = tid & 31;
  const int g = lane >> 4, ln = lane & 15;
  const int bh = blockIdx.y;             // 0..31
  const int b = bh >> 4, h = bh & 15;
  const size_t head = (size_t)bh * 2048 * 64;
  const _Float16* Q = qf + head;
  const _Float16* Km = kf + head;
  const _Float16* Vm = vf + head;
  const int q0 = blockIdx.x * 128 + w * 16;

  // Q fragments (16 rows x 64 d) straight from global (rows are 128B).
  const _Float16* qrow = Q + (size_t)(q0 + ln) * 64;
  v16h qa0 = load_a_frag(qrow, g);         // d 0..31
  v16h qa1 = load_a_frag(qrow + 32, g);    // d 32..63

  v16h ones;
#pragma unroll
  for (int i = 0; i < 16; ++i) ones[i] = (_Float16)1.0f;

  v8f o[4], lacc;
  float mi[8];
#pragma unroll
  for (int t = 0; t < 4; ++t)
#pragma unroll
    for (int i = 0; i < 8; ++i) o[t][i] = 0.0f;
#pragma unroll
  for (int i = 0; i < 8; ++i) { lacc[i] = 0.0f; mi[i] = -1e30f; }

  for (int kt = 0; kt < 16; ++kt) {
    if (kt + 1 < 16) {  // prefetch next tile's K and V rows
      int r = tid & 127;
      const _Float16* nxt =
          ((tid & 128) ? Vm : Km) + (size_t)((kt + 1) * 128 + r) * 64;
      __builtin_prefetch(nxt, 0, 3);
    }
    __syncthreads();                       // previous tile's Vt reads done
    // Stage V^T tile: Vt[d][n] = V[kt*128+n][d]
#pragma unroll
    for (int r = 0; r < 4; ++r) {
      int idx = r * 256 + tid;             // 1024 * 8 halves = 128x64
      int n = idx >> 3, d0 = (idx & 7) * 8;
      v8h vv = *(const v8h*)(Vm + (size_t)(kt * 128 + n) * 64 + d0);
#pragma unroll
      for (int j = 0; j < 8; ++j) Vt[(d0 + j) * 136 + n] = vv[j];
    }
    __syncthreads();

    // S = Q * K^T for 8 key sub-tiles of 16 (K frag: column = key, k = d)
    v8f s[8];
#pragma unroll
    for (int nf = 0; nf < 8; ++nf) {
      const _Float16* krow =
          Km + (size_t)(kt * 128 + nf * 16 + ln) * 64 + g * 16;
      v16h kb0 = load_b_frag(krow);        // d 0..31
      v16h kb1 = load_b_frag(krow + 32);   // d 32..63
      v8f acc;
#pragma unroll
      for (int i = 0; i < 8; ++i) acc[i] = 0.0f;
      acc = wmma_f16(qa0, kb0, acc);
      acc = wmma_f16(qa1, kb1, acc);
      s[nf] = acc;
    }

    // Row max: element i belongs to row m = g*8+i; its 16 columns live in one
    // 16-lane row -> VALU permlane16 butterflies.
    float mnew[8], alpha[8];
#pragma unroll
    for (int i = 0; i < 8; ++i) {
      float r = fmaxf(fmaxf(fmaxf(s[0][i], s[1][i]), fmaxf(s[2][i], s[3][i])),
                      fmaxf(fmaxf(s[4][i], s[5][i]), fmaxf(s[6][i], s[7][i])));
      r = row16_max(r);
      mnew[i] = fmaxf(mi[i], r);
      alpha[i] = __expf(mi[i] - mnew[i]);
      mi[i] = mnew[i];
    }

    // P = exp(S - mnew) -> per-wave LDS tile (C layout -> A layout transpose).
    _Float16* Pw = Pl + w * 16 * 136;
#pragma unroll
    for (int nf = 0; nf < 8; ++nf)
#pragma unroll
      for (int i = 0; i < 8; ++i) {
        float p = __expf(s[nf][i] - mnew[i]);
        Pw[(g * 8 + i) * 136 + nf * 16 + ln] = (_Float16)p;
      }

    // Rescale accumulators (o and l) by alpha.
#pragma unroll
    for (int i = 0; i < 8; ++i) {
      lacc[i] *= alpha[i];
#pragma unroll
      for (int t = 0; t < 4; ++t) o[t][i] *= alpha[i];
    }

    // Same-wave LDS ops execute in order; just stop compiler reordering.
    asm volatile("" ::: "memory");

    // Re-read P in A layout; O += P*V ; l += P*ones.
    v16h pa[4];
#pragma unroll
    for (int ks = 0; ks < 4; ++ks)
      pa[ks] = load_a_frag(Pw + ln * 136 + ks * 32, g);
#pragma unroll
    for (int t = 0; t < 4; ++t)
#pragma unroll
      for (int ks = 0; ks < 4; ++ks) {
        v16h vb = load_b_frag(Vt + (t * 16 + ln) * 136 + ks * 32 + g * 16);
        o[t] = wmma_f16(pa[ks], vb, o[t]);
      }
#pragma unroll
    for (int ks = 0; ks < 4; ++ks) lacc = wmma_f16(pa[ks], ones, lacc);
  }

  // Normalize and write merged-head output [B][N][C] as f16.
#pragma unroll
  for (int i = 0; i < 8; ++i) {
    float inv = 1.0f / lacc[i];
    int row = q0 + g * 8 + i;
#pragma unroll
    for (int t = 0; t < 4; ++t) {
      int col = h * 64 + t * 16 + ln;
      ao[((size_t)(b * 2048 + row)) * 1024 + col] = (_Float16)(o[t][i] * inv);
    }
  }
}

// ---------------------------------------------------------------------------
extern "C" void kernel_launch(void* const* d_in, const int* in_sizes, int n_in,
                              void* d_out, int out_size, void* d_ws, size_t ws_size,
                              hipStream_t stream) {
  (void)in_sizes; (void)n_in; (void)out_size; (void)ws_size;
  const float* x      = (const float*)d_in[0];  // [2,2048,1024]
  const float* w_qkv  = (const float*)d_in[1];  // [3072,1024]
  const float* b_qkv  = (const float*)d_in[2];  // [3072]
  const float* w_proj = (const float*)d_in[3];  // [1024,1024]
  const float* b_proj = (const float*)d_in[4];  // [1024]
  float* out = (float*)d_out;                   // [2,2048,1024]

  // Workspace (f16): xh (aliased by ao after QKV), wqh, wph, q, k, v.
  const size_t n_x  = (size_t)2 * 2048 * 1024;   // 4,194,304
  const size_t n_wq = (size_t)3072 * 1024;       // 3,145,728
  const size_t n_wp = (size_t)1024 * 1024;       // 1,048,576
  const size_t per  = (size_t)2 * 16 * 2048 * 64;// 4,194,304
  _Float16* xh  = (_Float16*)d_ws;               // also 'ao' later
  _Float16* wqh = xh + n_x;
  _Float16* wph = wqh + n_wq;
  _Float16* qf  = wph + n_wp;
  _Float16* kf  = qf + per;
  _Float16* vf  = kf + per;
  _Float16* ao  = xh;  // x is dead after the QKV gemm

  dim3 blk(256);
  cvt_f32_to_f16_kernel<<<(int)(n_x / 4 + 255) / 256, blk, 0, stream>>>(
      x, xh, (int)(n_x / 4));
  cvt_f32_to_f16_kernel<<<(int)(n_wq / 4 + 255) / 256, blk, 0, stream>>>(
      w_qkv, wqh, (int)(n_wq / 4));
  cvt_f32_to_f16_kernel<<<(int)(n_wp / 4 + 255) / 256, blk, 0, stream>>>(
      w_proj, wph, (int)(n_wp / 4));

  // QKV: M=4096 (B*N), K=1024, Nout=3072; tiles 128x128
  gemm_f16_kernel<0><<<dim3(32, 24), blk, 0, stream>>>(
      xh, wqh, b_qkv, qf, kf, vf, nullptr, 1024, 3072);
  // Attention: 16 query blocks x 32 (b,h)
  flash_attn_kernel<<<dim3(16, 32), blk, 0, stream>>>(qf, kf, vf, ao);
  // Proj: M=4096, K=1024, Nout=1024, f32 out; tiles 128x128
  gemm_f16_kernel<1><<<dim3(32, 8), blk, 0, stream>>>(
      ao, wph, b_proj, nullptr, nullptr, nullptr, out, 1024, 1024);
}
